// SparseUNetWithAttention_38843684225425
// MI455X (gfx1250) — compile-verified, matrix operand order checked by reading, not run on hardware
//
#include <hip/hip_runtime.h>
#include <hip/hip_bf16.h>
#include <math.h>

typedef __attribute__((ext_vector_type(16))) _Float16 v16h;
typedef __attribute__((ext_vector_type(8)))  _Float16 v8h;
typedef __attribute__((ext_vector_type(8)))  float    v8f;

#define TILE_K 32
#define BM 32          // block M tile (2 waves)
#define BN 64          // block N tile (2 waves x 2 accumulators)

// ---------------------------------------------------------------------------
// Implicit-GEMM conv / transposed-conv / 1x1-GEMM via WMMA f16, f32 accum.
//   KS   = kernel size (1 or 3), compile-time -> constant div/mod
//   MODE = 0: conv (stride/pad runtime), 1: stride-2 transposed conv (p_lo=2)
// Both LDS tiles are K-contiguous per row/col so every WMMA fragment is two
// 16-byte ds_load_b128's; A-staging decodes once per aligned 8-wide K-run
// (a run never crosses a tap boundary since Ci>=32 is a power of two) and
// uses two global float4 loads + one ds_store_b128.
// Block: 128 threads = 4 waves -> 32(M) x 64(N) tile, 2 WMMA per wave/K-step.
// ---------------------------------------------------------------------------
template<int KS, int MODE>
__global__ void conv3d_wmma(const float* __restrict__ in, const float* __restrict__ wt,
                            const float* __restrict__ bias, const float* __restrict__ res,
                            float* __restrict__ out,
                            int Din, int ciShift, int doutShift, int Co,
                            int stride, int pad)
{
    __shared__ _Float16 lA[BM * TILE_K];   // [m][k], k contiguous : 2 KB
    __shared__ _Float16 lB[BN * TILE_K];   // [n][k], k contiguous : 4 KB

    const int tid  = threadIdx.x;          // 0..127
    const int lane = tid & 31;
    const int wave = tid >> 5;
    const int wm   = wave & 1;             // M half
    const int wn   = wave >> 1;            // N half
    const int m0   = blockIdx.x * BM;
    const int n0   = blockIdx.y * BN;

    const int Ci    = 1 << ciShift;
    const int Dout  = 1 << doutShift;
    const int M     = Dout * Dout * Dout;
    const int K     = KS * KS * KS * Ci;
    const int dMask = Dout - 1;

    // A staging: thread -> row r = tid&31, K-run start = (tid>>5)*8
    const int ar_r = tid & 31;
    const int ar_k = (tid >> 5) * 8;
    const int am   = m0 + ar_r;
    const int aox  = am & dMask;
    const int aoy  = (am >> doutShift) & dMask;
    const int aoz  = am >> (2 * doutShift);

    // B staging: thread -> col n = tid&63, K-half start = (tid>>6)*16
    const int  bn_n   = tid & 63;
    const int  bn_k   = (tid >> 6) * 16;
    const bool bvalid = (n0 + bn_n) < Co;

    v8f acc0 = {};   // cols [wn*32      , wn*32 + 16)
    v8f acc1 = {};   // cols [wn*32 + 16 , wn*32 + 32)

    for (int k0 = 0; k0 < K; k0 += TILE_K) {
        if (k0 + TILE_K < K && tid == 0)
            __builtin_prefetch(&wt[(size_t)(k0 + TILE_K) * Co + n0], 0, 1);

        // ---- stage A: decode once per 8-run, vector load + convert ----
        {
            int k   = k0 + ar_k;
            int ci  = k & (Ci - 1);
            int tap = k >> ciShift;          // constant div/mod by KS below
            int dx  = tap % KS; int t2 = tap / KS;
            int dy  = t2 % KS;  int dz = t2 / KS;
            int ix, iy, iz; bool ok = (am < M);
            if (MODE == 0) {
                iz = aoz * stride - pad + dz;
                iy = aoy * stride - pad + dy;
                ix = aox * stride - pad + dx;
                ok = ok && (unsigned)iz < (unsigned)Din &&
                           (unsigned)iy < (unsigned)Din &&
                           (unsigned)ix < (unsigned)Din;
            } else {                          // transposed, stride 2, k=3, p_lo=2
                int tz = aoz + dz - 2, ty = aoy + dy - 2, tx = aox + dx - 2;
                ok = ok && (tz >= 0) && (ty >= 0) && (tx >= 0) &&
                     !((tz | ty | tx) & 1);
                iz = tz >> 1; iy = ty >> 1; ix = tx >> 1;
                ok = ok && iz < Din && iy < Din && ix < Din;
            }
            v8h pk = {};
            if (ok) {
                const float4* p = (const float4*)(in +
                    (((size_t)((iz * Din + iy) * Din + ix)) << ciShift) + ci);
                float4 x0 = p[0], x1 = p[1];
                pk[0] = (_Float16)x0.x; pk[1] = (_Float16)x0.y;
                pk[2] = (_Float16)x0.z; pk[3] = (_Float16)x0.w;
                pk[4] = (_Float16)x1.x; pk[5] = (_Float16)x1.y;
                pk[6] = (_Float16)x1.z; pk[7] = (_Float16)x1.w;
            }
            *(v8h*)&lA[ar_r * TILE_K + ar_k] = pk;
        }
        // ---- stage B: 16 K's for one N per thread, stored K-contiguous ----
        {
            v8h p0 = {}, p1 = {};
            if (bvalid) {
                const float* wp = wt + (size_t)(k0 + bn_k) * Co + n0 + bn_n;
                #pragma unroll
                for (int j = 0; j < 8; ++j) {
                    p0[j] = (_Float16)wp[(size_t)j * Co];
                    p1[j] = (_Float16)wp[(size_t)(j + 8) * Co];
                }
            }
            *(v8h*)&lB[bn_n * TILE_K + bn_k]     = p0;
            *(v8h*)&lB[bn_n * TILE_K + bn_k + 8] = p1;
        }
        __syncthreads();

        // ---- fragments: each is two contiguous 16B LDS loads ----
        // lane L: row/col = L&15 ; K halves: L<16 -> {0..7,16..23}, else {8..15,24..31}
        const int rc = lane & 15;
        const int kb = (lane >> 4) * 8;
        v8h alo  = *(const v8h*)&lA[(wm * 16 + rc) * TILE_K + kb];
        v8h ahi  = *(const v8h*)&lA[(wm * 16 + rc) * TILE_K + 16 + kb];
        v8h b0lo = *(const v8h*)&lB[(wn * 32 + rc) * TILE_K + kb];
        v8h b0hi = *(const v8h*)&lB[(wn * 32 + rc) * TILE_K + 16 + kb];
        v8h b1lo = *(const v8h*)&lB[(wn * 32 + 16 + rc) * TILE_K + kb];
        v8h b1hi = *(const v8h*)&lB[(wn * 32 + 16 + rc) * TILE_K + 16 + kb];
        v16h a, b0, b1;
        #pragma unroll
        for (int i = 0; i < 8; ++i) {
            a[i]  = alo[i];  a[i + 8]  = ahi[i];
            b0[i] = b0lo[i]; b0[i + 8] = b0hi[i];
            b1[i] = b1lo[i]; b1[i + 8] = b1hi[i];
        }
        acc0 = __builtin_amdgcn_wmma_f32_16x16x32_f16(false, a, false, b0, (short)0, acc0, false, false);
        acc1 = __builtin_amdgcn_wmma_f32_16x16x32_f16(false, a, false, b1, (short)0, acc1, false, false);
        __syncthreads();
    }

    // ---- store: VGPR r holds M=r (lanes 0-15) / M=r+8 (lanes 16-31), N=lane&15
    const int mbase = m0 + wm * 16 + ((lane >> 4) * 8);
    const int na = n0 + wn * 32 + (lane & 15);
    const int nb = na + 16;
    #pragma unroll
    for (int r = 0; r < 8; ++r) {
        int m = mbase + r;
        if (m < M) {
            if (na < Co) {
                float v = acc0[r];
                if (bias) v += bias[na];
                if (res)  v += res[(size_t)m * Co + na];
                out[(size_t)m * Co + na] = v;
            }
            if (nb < Co) {
                float v = acc1[r];
                if (bias) v += bias[nb];
                if (res)  v += res[(size_t)m * Co + nb];
                out[(size_t)m * Co + nb] = v;
            }
        }
    }
}

// ---------------------------------------------------------------------------
// BatchNorm stats: block handles 64 consecutive channels, lane-contiguous
// global reads; LDS tree over the 4 row-groups.
// ---------------------------------------------------------------------------
__global__ void bn_stats(const float* __restrict__ x, int M, int C,
                         float* __restrict__ mean, float* __restrict__ istd)
{
    __shared__ float s1[256], s2[256];
    const int clane = threadIdx.x & 63;
    const int mg    = threadIdx.x >> 6;        // row group 0..3
    const int c     = blockIdx.x * 64 + clane;
    float a = 0.0f, b = 0.0f;
    if (c < C) {
        for (int m = mg; m < M; m += 4) {
            float v = x[(size_t)m * C + c];
            a += v; b += v * v;
        }
    }
    s1[threadIdx.x] = a; s2[threadIdx.x] = b;
    __syncthreads();
    if (threadIdx.x < 128) {
        s1[threadIdx.x] += s1[threadIdx.x + 128];
        s2[threadIdx.x] += s2[threadIdx.x + 128];
    }
    __syncthreads();
    if (threadIdx.x < 64) {
        float sa = s1[threadIdx.x] + s1[threadIdx.x + 64];
        float sb = s2[threadIdx.x] + s2[threadIdx.x + 64];
        if (c < C) {
            float mu  = sa / (float)M;
            float var = sb / (float)M - mu * mu;
            mean[c] = mu;
            istd[c] = rsqrtf(var + 1e-5f);
        }
    }
}

// C is a power of two -> cShift/mask
__global__ void bn_apply(const float* __restrict__ x, float* __restrict__ y,
                         const float* __restrict__ mean, const float* __restrict__ istd,
                         const float* __restrict__ g, const float* __restrict__ b,
                         size_t total, int cShift, int relu)
{
    size_t i = (size_t)blockIdx.x * blockDim.x + threadIdx.x;
    if (i >= total) return;
    int c = (int)(i & (((size_t)1 << cShift) - 1));
    float v = (x[i] - mean[c]) * istd[c] * g[c] + b[c];
    if (relu) v = fmaxf(v, 0.0f);
    y[i] = v;
}

// Channel concat (C power of two): out[m,0:C]=a, out[m,C:2C]=b
__global__ void concat2(const float* __restrict__ a, const float* __restrict__ b,
                        float* __restrict__ o, size_t total, int cShift)
{
    size_t i = (size_t)blockIdx.x * blockDim.x + threadIdx.x;
    if (i >= total) return;
    int    C = 1 << cShift;
    size_t m = i >> (cShift + 1);
    int    c = (int)(i & (size_t)(2 * C - 1));
    o[i] = (c < C) ? a[(m << cShift) + c] : b[(m << cShift) + (c - C)];
}

// ---------------------------------------------------------------------------
// Single-head SDPA over N voxels (compile-time N, C): scores, softmax, P @ V.
// ---------------------------------------------------------------------------
template<int N, int C>
__global__ void attn_core(const float* __restrict__ qkv, float* __restrict__ o)
{
    __shared__ float sc[N * N];
    const int t = threadIdx.x;
    if (t < N * N) {
        int i = t / N, j = t % N;
        float s = 0.0f;
        for (int c = 0; c < C; ++c)
            s += qkv[(size_t)i * 3 * C + c] * qkv[(size_t)j * 3 * C + C + c];
        sc[t] = s / sqrtf((float)C);
    }
    __syncthreads();
    if (t < N) {
        float mx = -1e30f;
        for (int j = 0; j < N; ++j) mx = fmaxf(mx, sc[t * N + j]);
        float sum = 0.0f;
        for (int j = 0; j < N; ++j) { float e = __expf(sc[t * N + j] - mx); sc[t * N + j] = e; sum += e; }
        for (int j = 0; j < N; ++j) sc[t * N + j] /= sum;
    }
    __syncthreads();
    for (int idx = t; idx < N * C; idx += blockDim.x) {
        int i = idx / C, c = idx % C;
        float s = 0.0f;
        for (int j = 0; j < N; ++j)
            s += sc[i * N + j] * qkv[(size_t)j * 3 * C + 2 * C + c];
        o[idx] = s;
    }
}

// ---------------------------------------------------------------------------
// Host-side helpers
// ---------------------------------------------------------------------------
static inline int ilog2(int v) { return 31 - __builtin_clz(v); }

static void run_conv(hipStream_t s, const float* in, const float* w,
                     const float* bias, const float* res, float* out,
                     int Din, int Ci, int Dout, int Co,
                     int ks, int stride, int pad, int mode)
{
    int M = Dout * Dout * Dout;
    dim3 grid((unsigned)((M + BM - 1) / BM), (unsigned)((Co + BN - 1) / BN));
    int ciS = ilog2(Ci), doS = ilog2(Dout);
    if (mode == 1)
        conv3d_wmma<3, 1><<<grid, 128, 0, s>>>(in, w, bias, res, out, Din, ciS, doS, Co, 2, 0);
    else if (ks == 3)
        conv3d_wmma<3, 0><<<grid, 128, 0, s>>>(in, w, bias, res, out, Din, ciS, doS, Co, stride, pad);
    else
        conv3d_wmma<1, 0><<<grid, 128, 0, s>>>(in, w, bias, res, out, Din, ciS, doS, Co, stride, pad);
}

static void run_bn(hipStream_t s, const float* x, float* y,
                   const float* g, const float* b, float* mean, float* istd,
                   int M, int C, int relu)
{
    bn_stats<<<(C + 63) / 64, 256, 0, s>>>(x, M, C, mean, istd);
    size_t n = (size_t)M * C;
    bn_apply<<<(unsigned)((n + 255) / 256), 256, 0, s>>>(x, y, mean, istd, g, b, n, ilog2(C), relu);
}

static void run_concat(hipStream_t s, const float* a, const float* b, float* o,
                       int M, int C)
{
    size_t n = (size_t)M * 2 * C;
    concat2<<<(unsigned)((n + 255) / 256), 256, 0, s>>>(a, b, o, n, ilog2(C));
}

extern "C" void kernel_launch(void* const* d_in, const int* in_sizes, int n_in,
                              void* d_out, int out_size, void* d_ws, size_t ws_size,
                              hipStream_t stream)
{
    (void)in_sizes; (void)n_in; (void)out_size; (void)ws_size;

    const float* x     = (const float*)d_in[0];
    const float* e_w[4] = {(const float*)d_in[1],  (const float*)d_in[4],
                           (const float*)d_in[7],  (const float*)d_in[10]};
    const float* e_g[4] = {(const float*)d_in[2],  (const float*)d_in[5],
                           (const float*)d_in[8],  (const float*)d_in[11]};
    const float* e_b[4] = {(const float*)d_in[3],  (const float*)d_in[6],
                           (const float*)d_in[9],  (const float*)d_in[12]};
    const float* b0_w = (const float*)d_in[13];
    const float* b0_g = (const float*)d_in[14];
    const float* b0_b = (const float*)d_in[15];
    const float* at_g = (const float*)d_in[16];
    const float* at_b = (const float*)d_in[17];
    const float* at_qw = (const float*)d_in[18];
    const float* at_qb = (const float*)d_in[19];
    const float* at_pw = (const float*)d_in[20];
    const float* at_pb = (const float*)d_in[21];
    const float* b1_w = (const float*)d_in[22];
    const float* b1_g = (const float*)d_in[23];
    const float* b1_b = (const float*)d_in[24];
    const float* d_uw[4] = {(const float*)d_in[25], (const float*)d_in[29],
                            (const float*)d_in[33], (const float*)d_in[37]};
    const float* d_g[4]  = {(const float*)d_in[26], (const float*)d_in[30],
                            (const float*)d_in[34], (const float*)d_in[38]};
    const float* d_b[4]  = {(const float*)d_in[27], (const float*)d_in[31],
                            (const float*)d_in[35], (const float*)d_in[39]};
    const float* d_cw[4] = {(const float*)d_in[28], (const float*)d_in[32],
                            (const float*)d_in[36], (const float*)d_in[40]};
    const float* f_w = (const float*)d_in[41];
    const float* f_g = (const float*)d_in[42];
    const float* f_b = (const float*)d_in[43];

    // ---- workspace partition (fp32 elements) ----
    float* ws = (float*)d_ws;
    size_t off = 0;
    auto alloc = [&](size_t n) { float* p = ws + off; off += n; return p; };

    float* s0   = alloc((size_t)32768 * 64);
    float* s1   = alloc((size_t)4096 * 128);
    float* s2   = alloc((size_t)512 * 256);
    float* s3   = alloc((size_t)64 * 512);
    float* h0   = alloc((size_t)8 * 1024);
    float* fn   = alloc((size_t)8 * 1024);
    float* qkv  = alloc((size_t)8 * 3072);
    float* ao   = alloc((size_t)8 * 1024);
    float* h1   = alloc((size_t)8 * 1024);
    float* h2   = alloc((size_t)8 * 1024);
    float* u0   = alloc((size_t)64 * 512);
    float* cat0 = alloc((size_t)64 * 1024);
    float* c0   = alloc((size_t)64 * 512);
    float* u1   = alloc((size_t)512 * 256);
    float* cat1 = alloc((size_t)512 * 512);
    float* c1   = alloc((size_t)512 * 256);
    float* u2   = alloc((size_t)4096 * 128);
    float* cat2 = alloc((size_t)4096 * 256);
    float* c2   = alloc((size_t)4096 * 128);
    float* u3   = alloc((size_t)32768 * 64);
    float* cat3 = alloc((size_t)32768 * 128);
    float* c3   = alloc((size_t)32768 * 64);
    float* fin  = alloc((size_t)262144 * 32);
    float* mean = alloc(2048);
    float* istd = alloc(2048);

    // ---- encoder: 4x (stride-2 conv + BN + ReLU) ----
    run_conv(stream, x,  e_w[0], nullptr, nullptr, s0, 64, 32, 32, 64,  3, 2, 0, 0);
    run_bn(stream, s0, s0, e_g[0], e_b[0], mean, istd, 32768, 64, 1);
    run_conv(stream, s0, e_w[1], nullptr, nullptr, s1, 32, 64, 16, 128, 3, 2, 0, 0);
    run_bn(stream, s1, s1, e_g[1], e_b[1], mean, istd, 4096, 128, 1);
    run_conv(stream, s1, e_w[2], nullptr, nullptr, s2, 16, 128, 8, 256, 3, 2, 0, 0);
    run_bn(stream, s2, s2, e_g[2], e_b[2], mean, istd, 512, 256, 1);
    run_conv(stream, s2, e_w[3], nullptr, nullptr, s3, 8, 256, 4, 512,  3, 2, 0, 0);
    run_bn(stream, s3, s3, e_g[3], e_b[3], mean, istd, 64, 512, 1);

    // ---- bottleneck: stride-2 conv -> attention -> stride-1 conv ----
    run_conv(stream, s3, b0_w, nullptr, nullptr, h0, 4, 512, 2, 1024, 3, 2, 0, 0);
    run_bn(stream, h0, h0, b0_g, b0_b, mean, istd, 8, 1024, 1);

    run_bn(stream, h0, fn, at_g, at_b, mean, istd, 8, 1024, 0);          // BN (no relu)
    run_conv(stream, fn, at_qw, at_qb, nullptr, qkv, 2, 1024, 2, 3072, 1, 1, 0, 0); // qkv
    attn_core<8, 1024><<<1, 256, 0, stream>>>(qkv, ao);                  // SDPA
    run_conv(stream, ao, at_pw, at_pb, h0, h1, 2, 1024, 2, 1024, 1, 1, 0, 0);       // proj + residual

    run_conv(stream, h1, b1_w, nullptr, nullptr, h2, 2, 1024, 2, 1024, 3, 1, 1, 0);
    run_bn(stream, h2, h2, b1_g, b1_b, mean, istd, 8, 1024, 1);

    // ---- decoder: 4x (upconv + BN + ReLU, concat skip, 1x1 conv) ----
    run_conv(stream, h2, d_uw[0], nullptr, nullptr, u0, 2, 1024, 4, 512, 3, 2, 0, 1);
    run_bn(stream, u0, u0, d_g[0], d_b[0], mean, istd, 64, 512, 1);
    run_concat(stream, u0, s3, cat0, 64, 512);
    run_conv(stream, cat0, d_cw[0], nullptr, nullptr, c0, 4, 1024, 4, 512, 1, 1, 0, 0);

    run_conv(stream, c0, d_uw[1], nullptr, nullptr, u1, 4, 512, 8, 256, 3, 2, 0, 1);
    run_bn(stream, u1, u1, d_g[1], d_b[1], mean, istd, 512, 256, 1);
    run_concat(stream, u1, s2, cat1, 512, 256);
    run_conv(stream, cat1, d_cw[1], nullptr, nullptr, c1, 8, 512, 8, 256, 1, 1, 0, 0);

    run_conv(stream, c1, d_uw[2], nullptr, nullptr, u2, 8, 256, 16, 128, 3, 2, 0, 1);
    run_bn(stream, u2, u2, d_g[2], d_b[2], mean, istd, 4096, 128, 1);
    run_concat(stream, u2, s1, cat2, 4096, 128);
    run_conv(stream, cat2, d_cw[2], nullptr, nullptr, c2, 16, 256, 16, 128, 1, 1, 0, 0);

    run_conv(stream, c2, d_uw[3], nullptr, nullptr, u3, 16, 128, 32, 64, 3, 2, 0, 1);
    run_bn(stream, u3, u3, d_g[3], d_b[3], mean, istd, 32768, 64, 1);
    run_concat(stream, u3, s0, cat3, 32768, 64);
    run_conv(stream, cat3, d_cw[3], nullptr, nullptr, c3, 32, 128, 32, 64, 1, 1, 0, 0);

    // ---- final: upconv + BN + ReLU -> d_out (64^3 x 32 fp32) ----
    run_conv(stream, c3, f_w, nullptr, nullptr, fin, 32, 64, 64, 32, 3, 2, 0, 1);
    run_bn(stream, fin, (float*)d_out, f_g, f_b, mean, istd, 262144, 32, 1);
}
